// RPN_15109694947635
// MI455X (gfx1250) — matrix operand order repaired and checked by reading, not compiled, not run on hardware
//
#include <hip/hip_runtime.h>
#include <hip/hip_bf16.h>

// ---------------------------------------------------------------------------
// RPN forward for MI455X (gfx1250): 3x3 conv + 1x1 heads as BF16 WMMA GEMMs,
// softmax/decode/top-k/NMS as supporting kernels.
// Conv uses a halo-padded NHWC bf16 feature map so the WMMA inner loop has
// zero control flow: 12 unconditional b128 loads -> 8 v_wmma per k-step.
// ---------------------------------------------------------------------------

typedef __attribute__((ext_vector_type(16))) __bf16 bf16x16;
typedef __attribute__((ext_vector_type(8)))  __bf16 bf16x8;
typedef __attribute__((ext_vector_type(8)))  float  f32x8;

#define BATCH   2
#define CIN     512
#define HH      100
#define WW      152
#define HP      102        // padded height
#define WP      154        // padded width
#define PPAD    15708      // HP*WP
#define HWPIX   15200      // 100*152
#define KTOT    4608       // 9*512
#define NANCH   9
#define NPROP   136800     // HWPIX*9
#define PRE_NMS 6000
#define POST_NMS 300

__device__ __forceinline__ unsigned keyOf(float s) {
    unsigned u = __float_as_uint(s);
    return (u & 0x80000000u) ? ~u : (u | 0x80000000u);
}

// ------------- prep: NCHW f32 -> halo-padded NHWC bf16 feature map -----------
__global__ void k_featP(const float* __restrict__ bf, __bf16* __restrict__ featP) {
    int i = blockIdx.x * 256 + threadIdx.x;
    if (i >= BATCH * PPAD * CIN) return;
    int n  = i / (PPAD * CIN);
    int r  = i - n * (PPAD * CIN);
    int pp = r / CIN;
    int c  = r - pp * CIN;
    int yy = pp / WP, xx = pp - yy * WP;
    int y = yy - 1, x = xx - 1;
    float v = 0.f;
    if (y >= 0 && y < HH && x >= 0 && x < WW)
        v = bf[(n * CIN + c) * HWPIX + y * WW + x];
    featP[i] = (__bf16)v;
}

// weights OIHW f32 -> [o][rs][c] bf16 (K contiguous per output row)
__global__ void k_wb(const float* __restrict__ W, __bf16* __restrict__ Wb) {
    int i = blockIdx.x * 256 + threadIdx.x;
    if (i >= CIN * KTOT) return;
    int o  = i / KTOT;
    int r  = i - o * KTOT;
    int rs = r / CIN;
    int c  = r - rs * CIN;
    Wb[i] = (__bf16)W[(o * CIN + c) * 9 + rs];
}

// combined 1x1 weights: rows 0..17 = cls, 18..53 = reg, 54..63 = zero pad
__global__ void k_wc(const float* __restrict__ Wcls, const float* __restrict__ Wreg,
                     __bf16* __restrict__ Wc) {
    int i = blockIdx.x * 256 + threadIdx.x;
    if (i >= 64 * CIN) return;
    int co = i / CIN;
    int c  = i - co * CIN;
    float v = 0.f;
    if (co < 18)      v = Wcls[co * CIN + c];
    else if (co < 54) v = Wreg[(co - 18) * CIN + c];
    Wc[i] = (__bf16)v;
}

// ---------------- 3x3 conv as implicit GEMM with BF16 WMMA -------------------
// wave tile: 32 output channels x 64 pixels; 9*16*8 wmma per wave
#define NT64 238   // ceil(950/4) 64-pixel tiles
#define NMO2 16    // 512/32 channel groups
__global__ void k_conv(const __bf16* __restrict__ featP,
                       const __bf16* __restrict__ Wb,
                       const float*  __restrict__ bias,
                       __bf16* __restrict__ xfeat) {
    int w = blockIdx.x * 8 + (threadIdx.x >> 5);
    const int totalW = BATCH * NMO2 * NT64;
    if (w >= totalW) return;
    int lane   = threadIdx.x & 31;
    int laneLo = lane & 15;
    int hi     = lane >> 4;

    int n   = w / (NMO2 * NT64);
    int r0  = w - n * (NMO2 * NT64);
    int mo2 = r0 / NT64;
    int nt  = r0 - mo2 * NT64;
    int P0  = nt * 64;

    const __bf16* wr0 = Wb + (mo2 * 32 + laneLo) * KTOT;       // rows +0..15
    const __bf16* wr1 = wr0 + 16 * KTOT;                        // rows +16..31

    f32x8 acc[2][4];
    #pragma unroll
    for (int h = 0; h < 2; ++h)
        for (int t = 0; t < 4; ++t)
            for (int j = 0; j < 8; ++j) acc[h][t][j] = 0.f;

    int  px[4];
    bool pv[4];
    #pragma unroll
    for (int t = 0; t < 4; ++t) {
        int pb = P0 + t * 16;
        pv[t] = pb < HWPIX;
        px[t] = pv[t] ? (pb + laneLo) : laneLo;  // clamp: loads stay valid
    }

    for (int rs = 0; rs < 9; ++rs) {
        int dr = rs / 3, ds = rs % 3;            // padded coords: +0..2
        int boff[4];
        #pragma unroll
        for (int t = 0; t < 4; ++t) {
            int p = px[t];
            int y = p / WW, x = p - y * WW;
            boff[t] = ((n * PPAD + (y + dr) * WP + (x + ds)) * CIN) + hi * 16;
        }
        const __bf16* wa = wr0 + rs * CIN;
        const __bf16* wb = wr1 + rs * CIN;
        #pragma unroll 2
        for (int cb = 0; cb < 16; ++cb) {
            int c0 = cb * 32;
            bf16x8 a0l = *(const bf16x8*)(wa + c0 + hi * 16);
            bf16x8 a0h = *(const bf16x8*)(wa + c0 + 8 + hi * 16);
            bf16x8 a1l = *(const bf16x8*)(wb + c0 + hi * 16);
            bf16x8 a1h = *(const bf16x8*)(wb + c0 + 8 + hi * 16);
            bf16x16 a0, a1;
            #pragma unroll
            for (int i = 0; i < 8; ++i) {
                a0[i] = a0l[i]; a0[8 + i] = a0h[i];
                a1[i] = a1l[i]; a1[8 + i] = a1h[i];
            }
            #pragma unroll
            for (int t = 0; t < 4; ++t) {
                bf16x16 b = *(const bf16x16*)(featP + boff[t] + c0);
                acc[0][t] = __builtin_amdgcn_wmma_f32_16x16x32_bf16(
                    false, a0, false, b, (short)0, acc[0][t], false, false);
                acc[1][t] = __builtin_amdgcn_wmma_f32_16x16x32_bf16(
                    false, a1, false, b, (short)0, acc[1][t], false, false);
            }
        }
    }

    #pragma unroll
    for (int h = 0; h < 2; ++h) {
        int chB = mo2 * 32 + h * 16 + hi * 8;
        #pragma unroll
        for (int t = 0; t < 4; ++t) {
            if (!pv[t]) continue;
            bf16x8 o;
            #pragma unroll
            for (int j = 0; j < 8; ++j) {
                float v = acc[h][t][j] + bias[chB + j];
                o[j] = (__bf16)(v > 0.f ? v : 0.f);
            }
            *(bf16x8*)(xfeat + (n * HWPIX + px[t]) * CIN + chB) = o;
        }
    }
}

// ---------------- 1x1 heads (cls 18 + reg 36, padded to 64) via WMMA ---------
__global__ void k_head(const __bf16* __restrict__ xfeat, const __bf16* __restrict__ Wc,
                       const float* __restrict__ bcls, const float* __restrict__ breg,
                       float* __restrict__ out64) {
    int w = blockIdx.x * 8 + (threadIdx.x >> 5);
    if (w >= BATCH * 950) return;
    int lane = threadIdx.x & 31, laneLo = lane & 15, hi = lane >> 4;
    int n = w / 950, pt = w - n * 950;
    int p = pt * 16 + laneLo;

    const __bf16* bbase = xfeat + (n * HWPIX + p) * CIN + hi * 16;

    f32x8 acc[4];
    #pragma unroll
    for (int m = 0; m < 4; ++m)
        for (int j = 0; j < 8; ++j) acc[m][j] = 0.f;

    #pragma unroll 2
    for (int kb = 0; kb < 16; ++kb) {
        bf16x16 b = *(const bf16x16*)(bbase + kb * 32);
        #pragma unroll
        for (int m = 0; m < 4; ++m) {
            const __bf16* wr = Wc + (m * 16 + laneLo) * CIN + kb * 32 + hi * 16;
            bf16x8 a0 = *(const bf16x8*)wr;
            bf16x8 a1 = *(const bf16x8*)(wr + 8);
            bf16x16 a;
            #pragma unroll
            for (int i = 0; i < 8; ++i) { a[i] = a0[i]; a[8 + i] = a1[i]; }
            acc[m] = __builtin_amdgcn_wmma_f32_16x16x32_bf16(
                false, a, false, b, (short)0, acc[m], false, false);
        }
    }
    #pragma unroll
    for (int m = 0; m < 4; ++m) {
        int chB = m * 16 + hi * 8;
        f32x8 o;
        #pragma unroll
        for (int j = 0; j < 8; ++j) {
            int ch = chB + j;
            float bb = (ch < 18) ? bcls[ch] : (ch < 54 ? breg[ch - 18] : 0.f);
            o[j] = acc[m][j] + bb;
        }
        *(f32x8*)(out64 + (n * HWPIX + p) * 64 + chB) = o;
    }
}

// ---------------- softmax + box decode + clip --------------------------------
__global__ void k_proposal(const float* __restrict__ out64,
                           float* __restrict__ scores, float* __restrict__ boxes) {
    int i = blockIdx.x * 256 + threadIdx.x;
    if (i >= BATCH * NPROP) return;
    int n = i / NPROP;
    int r = i - n * NPROP;
    int p = r / NANCH;
    int a = r - p * NANCH;
    const float* cell = out64 + (n * HWPIX + p) * 64;
    float score = 1.f / (1.f + expf(cell[a] - cell[9 + a]));
    float dx = cell[18 + a * 4 + 0], dy = cell[18 + a * 4 + 1];
    float dw = cell[18 + a * 4 + 2], dh = cell[18 + a * 4 + 3];

    const float wsr[3] = {23.f, 16.f, 11.f};
    const float hsr[3] = {12.f, 16.f, 22.f};
    const float scl[3] = {8.f, 16.f, 32.f};
    float aw = wsr[a / 3] * scl[a % 3];
    float ah = hsr[a / 3] * scl[a % 3];
    int y = p / WW, x = p - y * WW;
    float sx = x * 16.f, sy = y * 16.f;
    float ax1 = sx + 7.5f - 0.5f * (aw - 1.f);
    float ay1 = sy + 7.5f - 0.5f * (ah - 1.f);
    float w0 = aw, h0 = ah;
    float cx = ax1 + 0.5f * w0, cy = ay1 + 0.5f * h0;
    float pcx = dx * w0 + cx, pcy = dy * h0 + cy;
    float pw = expf(dw) * w0, ph = expf(dh) * h0;
    float bx1 = fminf(fmaxf(pcx - 0.5f * pw, 0.f), 2431.f);
    float by1 = fminf(fmaxf(pcy - 0.5f * ph, 0.f), 1599.f);
    float bx2 = fminf(fmaxf(pcx + 0.5f * pw - 1.f, 0.f), 2431.f);
    float by2 = fminf(fmaxf(pcy + 0.5f * ph - 1.f, 0.f), 1599.f);
    scores[i] = score;
    float* bo = boxes + (size_t)i * 4;
    bo[0] = bx1; bo[1] = by1; bo[2] = bx2; bo[3] = by2;
}

// ---------------- top-6000 radix select --------------------------------------
__global__ void k_zero(unsigned* __restrict__ p, int nvals) {
    int i = blockIdx.x * 256 + threadIdx.x;
    if (i < nvals) p[i] = 0u;
}
__global__ void k_hist1(const float* __restrict__ scores, unsigned* __restrict__ hist) {
    int i = blockIdx.x * 256 + threadIdx.x;
    if (i >= BATCH * NPROP) return;
    int n = i / NPROP;
    atomicAdd(&hist[n * 65536 + (keyOf(scores[i]) >> 16)], 1u);
}
__global__ void k_scan1(const unsigned* __restrict__ hist, unsigned* __restrict__ misc) {
    if (threadIdx.x) return;
    int n = blockIdx.x;
    const unsigned* h = hist + n * 65536;
    unsigned cum = 0;
    for (int b = 65535; b >= 0; --b) {
        unsigned c = h[b];
        if (cum + c >= (unsigned)PRE_NMS) { misc[n * 4 + 0] = (unsigned)b; misc[n * 4 + 1] = cum; break; }
        cum += c;
    }
}
__global__ void k_hist2(const float* __restrict__ scores, const unsigned* __restrict__ misc,
                        unsigned* __restrict__ hist) {
    int i = blockIdx.x * 256 + threadIdx.x;
    if (i >= BATCH * NPROP) return;
    int n = i / NPROP;
    unsigned key = keyOf(scores[i]);
    if ((key >> 16) == misc[n * 4 + 0])
        atomicAdd(&hist[n * 65536 + (key & 0xFFFFu)], 1u);
}
__global__ void k_scan2(const unsigned* __restrict__ hist, unsigned* __restrict__ misc) {
    if (threadIdx.x) return;
    int n = blockIdx.x;
    const unsigned* h = hist + n * 65536;
    unsigned m = misc[n * 4 + 1];
    unsigned cum = 0;
    for (int b = 65535; b >= 0; --b) {
        unsigned c = h[b];
        if (m + cum + c >= (unsigned)PRE_NMS) { misc[n * 4 + 2] = (unsigned)b; misc[n * 4 + 3] = m + cum; break; }
        cum += c;
    }
}
__global__ void k_compact(const float* __restrict__ scores, const float* __restrict__ boxes,
                          unsigned* __restrict__ misc,
                          float* __restrict__ selS, float* __restrict__ selB) {
    int i = blockIdx.x * 256 + threadIdx.x;
    if (i >= BATCH * NPROP) return;
    int n = i / NPROP;
    unsigned key = keyOf(scores[i]);
    unsigned K32 = (misc[n * 4 + 0] << 16) | misc[n * 4 + 2];
    unsigned m2  = misc[n * 4 + 3];
    int pos = -1;
    if (key > K32) {
        pos = (int)atomicAdd(&misc[8 + 2 * n], 1u);
    } else if (key == K32) {
        unsigned e = atomicAdd(&misc[9 + 2 * n], 1u);
        unsigned q = m2 + e;
        if (q < (unsigned)PRE_NMS) pos = (int)q;
    }
    if (pos >= 0) {
        selS[n * PRE_NMS + pos] = scores[i];
        const float* bo = boxes + (size_t)i * 4;
        float* so = selB + (size_t)(n * PRE_NMS + pos) * 4;
        so[0] = bo[0]; so[1] = bo[1]; so[2] = bo[2]; so[3] = bo[3];
    }
}

// ---------------- greedy NMS (one workgroup per batch) -----------------------
__global__ void k_nms(const float* __restrict__ selS, const float* __restrict__ selB,
                      float* __restrict__ out) {
    __shared__ float ss[PRE_NMS];
    __shared__ float rv[1024];
    __shared__ int   ri[1024];
    int n = blockIdx.x, tid = threadIdx.x;
    for (int i = tid; i < PRE_NMS; i += 1024) ss[i] = selS[n * PRE_NMS + i];
    if (n == 0 && tid == 0) { out[3000] = 0.f; out[3001] = 0.f; }
    __syncthreads();
    const float NEG = -3.4e38f;
    for (int it = 0; it < POST_NMS; ++it) {
        float bv = NEG; int bi = 0x7fffffff;
        for (int i = tid; i < PRE_NMS; i += 1024) {
            float v = ss[i];
            if (v > bv || (v == bv && i < bi)) { bv = v; bi = i; }
        }
        rv[tid] = bv; ri[tid] = bi;
        __syncthreads();
        for (int off = 512; off > 0; off >>= 1) {
            if (tid < off) {
                float v2 = rv[tid + off]; int i2 = ri[tid + off];
                if (v2 > rv[tid] || (v2 == rv[tid] && i2 < ri[tid])) { rv[tid] = v2; ri[tid] = i2; }
            }
            __syncthreads();
        }
        int j = ri[0]; float vj = rv[0];
        bool ok = vj > NEG;
        float* orow = out + (n * POST_NMS + it) * 5;
        if (tid == 0) {
            orow[0] = (float)n;
            if (ok) {
                const float* bj = selB + (size_t)(n * PRE_NMS + j) * 4;
                orow[1] = bj[0]; orow[2] = bj[1]; orow[3] = bj[2]; orow[4] = bj[3];
            } else {
                orow[1] = orow[2] = orow[3] = orow[4] = 0.f;
            }
        }
        if (ok) {
            const float* bj = selB + (size_t)(n * PRE_NMS + j) * 4;
            float xj1 = bj[0], yj1 = bj[1], xj2 = bj[2], yj2 = bj[3];
            float areaj = (xj2 - xj1 + 1.f) * (yj2 - yj1 + 1.f);
            for (int i = tid; i < PRE_NMS; i += 1024) {
                if (ss[i] <= NEG) continue;
                const float* bb = selB + (size_t)(n * PRE_NMS + i) * 4;
                float xx1 = fmaxf(xj1, bb[0]), yy1 = fmaxf(yj1, bb[1]);
                float xx2 = fminf(xj2, bb[2]), yy2 = fminf(yj2, bb[3]);
                float iw = fmaxf(xx2 - xx1 + 1.f, 0.f);
                float ih = fmaxf(yy2 - yy1 + 1.f, 0.f);
                float inter = iw * ih;
                float areai = (bb[2] - bb[0] + 1.f) * (bb[3] - bb[1] + 1.f);
                float iou = inter / (areaj + areai - inter);
                if (iou > 0.7f) ss[i] = NEG;
            }
        }
        __syncthreads();
    }
}

// ---------------------------------------------------------------------------
extern "C" void kernel_launch(void* const* d_in, const int* in_sizes, int n_in,
                              void* d_out, int out_size, void* d_ws, size_t ws_size,
                              hipStream_t stream) {
    (void)in_sizes; (void)n_in; (void)out_size;
    const float* base_feat = (const float*)d_in[0];
    const float* W_inter   = (const float*)d_in[3];
    const float* b_inter   = (const float*)d_in[4];
    const float* W_cls     = (const float*)d_in[5];
    const float* b_cls     = (const float*)d_in[6];
    const float* W_reg     = (const float*)d_in[7];
    const float* b_reg     = (const float*)d_in[8];
    float* out = (float*)d_out;

    char* ws = (char*)d_ws;
    // persistent regions
    const size_t OFF_FEATP = 0;                                   // 32,169,984 B
    const size_t OFF_WB    = OFF_FEATP + (size_t)BATCH * PPAD * CIN * 2;
    const size_t OFF_XFEAT = OFF_WB    + (size_t)CIN * KTOT * 2;  // 4,718,592 B
    const size_t OFF_WC    = OFF_XFEAT + (size_t)BATCH * HWPIX * CIN * 2;
    const size_t TOTAL     = OFF_WC + (size_t)64 * CIN * 2;       // ~68 MB
    if (ws_size < TOTAL) return;
    // post-conv buffers alias the (dead) featP region
    const size_t O_OUT64 = 0;                       // 7,782,400 B
    const size_t O_SCORE = O_OUT64 + 7782400;       // 1,094,400 B
    const size_t O_BOXES = O_SCORE + 1094400;       // 4,377,600 B
    const size_t O_HIST  = O_BOXES + 4377600;       // 524,288 B
    const size_t O_MISC  = O_HIST  + 524288;        // 64 B
    const size_t O_SELS  = O_MISC  + 64;            // 48,000 B
    const size_t O_SELB  = O_SELS  + 48000;         // 192,000 B

    __bf16* featP = (__bf16*)(ws + OFF_FEATP);
    __bf16* Wb    = (__bf16*)(ws + OFF_WB);
    __bf16* xfeat = (__bf16*)(ws + OFF_XFEAT);
    __bf16* Wc    = (__bf16*)(ws + OFF_WC);
    float*  out64 = (float*)(ws + O_OUT64);
    float*  scor  = (float*)(ws + O_SCORE);
    float*  boxs  = (float*)(ws + O_BOXES);
    unsigned* hist = (unsigned*)(ws + O_HIST);
    unsigned* misc = (unsigned*)(ws + O_MISC);
    float*  selS  = (float*)(ws + O_SELS);
    float*  selB  = (float*)(ws + O_SELB);

    // prep
    k_featP<<<(BATCH * PPAD * CIN + 255) / 256, 256, 0, stream>>>(base_feat, featP);
    k_wb<<<(CIN * KTOT + 255) / 256, 256, 0, stream>>>(W_inter, Wb);
    k_wc<<<(64 * CIN + 255) / 256, 256, 0, stream>>>(W_cls, W_reg, Wc);
    // conv 3x3 (WMMA, 32x64 tile per wave)
    k_conv<<<(BATCH * NMO2 * NT64 + 7) / 8, 256, 0, stream>>>(featP, Wb, b_inter, xfeat);
    // 1x1 heads (WMMA); out64 overwrites dead featP region
    k_head<<<(BATCH * 950 + 7) / 8, 256, 0, stream>>>(xfeat, Wc, b_cls, b_reg, out64);
    // decode
    k_proposal<<<(BATCH * NPROP + 255) / 256, 256, 0, stream>>>(out64, scor, boxs);
    // top-6000 radix select
    k_zero<<<(BATCH * 65536 + 255) / 256, 256, 0, stream>>>(hist, BATCH * 65536);
    k_zero<<<1, 256, 0, stream>>>(misc, 16);
    k_hist1<<<(BATCH * NPROP + 255) / 256, 256, 0, stream>>>(scor, hist);
    k_scan1<<<BATCH, 32, 0, stream>>>(hist, misc);
    k_zero<<<(BATCH * 65536 + 255) / 256, 256, 0, stream>>>(hist, BATCH * 65536);
    k_hist2<<<(BATCH * NPROP + 255) / 256, 256, 0, stream>>>(scor, misc, hist);
    k_scan2<<<BATCH, 32, 0, stream>>>(hist, misc);
    k_compact<<<(BATCH * NPROP + 255) / 256, 256, 0, stream>>>(scor, boxs, misc, selS, selB);
    // NMS + final ROIs
    k_nms<<<BATCH, 1024, 0, stream>>>(selS, selB, out);
}
